// FlowControlRNN_4810363372394
// MI455X (gfx1250) — compile-verified, matrix operand order checked by reading, not run on hardware
//
#include <hip/hip_runtime.h>
#include <hip/hip_bf16.h>

typedef __attribute__((ext_vector_type(16))) __bf16 v16bf;
typedef __attribute__((ext_vector_type(8)))  __bf16 v8bf;
typedef __attribute__((ext_vector_type(8)))  float  v8f;
typedef __attribute__((ext_vector_type(4)))  float  vf4;

static constexpr int PADIDX = 1;
static constexpr int Bn = 64;    // batch
static constexpr int Sn = 512;   // seq len
static constexpr int En = 256;   // embed dim
static constexpr int Hn = 512;   // hidden dim
static constexpr int NWG = 16;   // persistent workgroups in recurrence

static __device__ inline v16bf cat16(v8bf lo, v8bf hi) {
    return __builtin_shufflevector(lo, hi, 0, 1, 2, 3, 4, 5, 6, 7,
                                           8, 9, 10, 11, 12, 13, 14, 15);
}

// ---------------------------------------------------------------------------
// Kernel 0: zero grid-sync counter; transpose W_h/W_r into bf16 [H][E] so the
// recurrence-free projection GEMM can read B fragments as contiguous 16B loads.
// ---------------------------------------------------------------------------
__global__ __launch_bounds__(256) void k_prep(
    const float* __restrict__ W_h, const float* __restrict__ W_r,
    __bf16* __restrict__ WhT, __bf16* __restrict__ WrT,
    unsigned* __restrict__ counter)
{
    if (blockIdx.x == 0 && threadIdx.x == 0) *counter = 0u;
    const int i = blockIdx.x * 256 + threadIdx.x;   // over E*H = 131072
    if (i < En * Hn) {
        const int k = i >> 9;        // row in [E]
        const int n = i & (Hn - 1);  // col in [H]
        WhT[n * En + k] = (__bf16)W_h[i];   // coalesced read, scattered 2B write
        WrT[n * En + k] = (__bf16)W_r[i];
    }
}

// ---------------------------------------------------------------------------
// Kernel 1: fused embedding gather + input projections, time-major output
// (row = s*64 + b).  One block per 16-row tile: the 16 embedding rows are
// gathered ONCE into LDS (bf16, padded stride), then 8 waves each produce 4
// of the 32 column tiles with 16x16x32 bf16 WMMA.
//   xh[row,n] = (emb[x] @ W_h)[n] + b_Wh[n] + b_h[n]
//   R [row,n] = sigmoid((emb[x] @ W_r)[n] + b_Wr[n] + b_r[n])
// ---------------------------------------------------------------------------
__global__ __launch_bounds__(256) void k_embed_project(
    const int*    __restrict__ x,
    const float*  __restrict__ emb,
    const __bf16* __restrict__ WhT, const __bf16* __restrict__ WrT,
    const float*  __restrict__ b_Wh, const float* __restrict__ b_h,
    const float*  __restrict__ b_Wr, const float* __restrict__ b_r,
    float* __restrict__ xh, float* __restrict__ Rg)
{
    constexpr int AS = En + 8;                 // padded row stride (bank skew)
    __shared__ __bf16 As[16 * AS];

    const int rt  = blockIdx.x;                // 2048 row tiles
    const int tid = threadIdx.x;

    // ---- Stage A tile: 16 rows x 256 cols, pad row -> zeros via multiply ----
    {
        const int r  = tid >> 4;               // row 0..15
        const int kb = (tid & 15) * 16;        // 16 consecutive k per thread
        const int row = rt * 16 + r;
        const int s = row >> 6, b = row & 63;
        const int tok = x[b * Sn + s];
        const float sc = (tok == PADIDX) ? 0.0f : 1.0f;   // no divergent loads
        const vf4* src = (const vf4*)(emb + (long)tok * En + kb);
        __bf16* dst = &As[r * AS + kb];
        vf4 v0 = src[0], v1 = src[1], v2 = src[2], v3 = src[3];
        v8bf p0, p1;
#pragma unroll
        for (int e = 0; e < 4; ++e) {
            p0[e]     = (__bf16)(v0[e] * sc);
            p0[e + 4] = (__bf16)(v1[e] * sc);
            p1[e]     = (__bf16)(v2[e] * sc);
            p1[e + 4] = (__bf16)(v3[e] * sc);
        }
        *(v8bf*)(dst)     = p0;                // 16B LDS stores (aligned)
        *(v8bf*)(dst + 8) = p1;
    }
    __syncthreads();

    const int lane = tid & 31;
    const int wave = tid >> 5;
    const int ml = lane & 15;
    const int kh = lane >> 4;                  // K-half covered by this lane
    const __bf16* a_row = &As[ml * AS];

#pragma unroll
    for (int q = 0; q < 4; ++q) {
        const int ct = wave * 4 + q;           // column tile 0..31
        const int n  = ct * 16 + ml;           // output column (C/D: N=lane%16)
        const __bf16* bh_row = WhT + (long)n * En;
        const __bf16* br_row = WrT + (long)n * En;

        v8f acc_h = {};
        v8f acc_r = {};
        for (int kc = 0; kc < En / 32; ++kc) {
            const int k0 = kc * 32 + kh * 8;
            const v16bf a  = cat16(*(const v8bf*)(a_row + k0),
                                   *(const v8bf*)(a_row + k0 + 16));
            const v16bf bh = cat16(*(const v8bf*)(bh_row + k0),
                                   *(const v8bf*)(bh_row + k0 + 16));
            const v16bf br = cat16(*(const v8bf*)(br_row + k0),
                                   *(const v8bf*)(br_row + k0 + 16));
            acc_h = __builtin_amdgcn_wmma_f32_16x16x32_bf16(
                false, a, false, bh, (short)0, acc_h, false, false);
            acc_r = __builtin_amdgcn_wmma_f32_16x16x32_bf16(
                false, a, false, br, (short)0, acc_r, false, false);
        }

        const float bias_h = b_Wh[n] + b_h[n];
        const float bias_r = b_Wr[n] + b_r[n];
#pragma unroll
        for (int r8 = 0; r8 < 8; ++r8) {
            const int rowg = rt * 16 + r8 + kh * 8;  // C/D: M = r + 8*(lane/16)
            const long idx = (long)rowg * Hn + n;
            xh[idx] = acc_h[r8] + bias_h;
            const float z = acc_r[r8] + bias_r;
            Rg[idx] = 1.0f / (1.0f + __expf(-z));
        }
    }
}

// ---------------------------------------------------------------------------
// Device-wide barrier for the persistent recurrence kernel.
// ---------------------------------------------------------------------------
__device__ inline void grid_sync(unsigned* counter, unsigned target) {
    __threadfence();          // release this thread's stores to device scope
    __syncthreads();
    if (threadIdx.x == 0) {
        atomicAdd(counter, 1u);
        while (__hip_atomic_load(counter, __ATOMIC_RELAXED,
                                 __HIP_MEMORY_SCOPE_AGENT) < target) {
            __builtin_amdgcn_s_sleep(1);
        }
    }
    __syncthreads();
    __threadfence();          // acquire
}

// ---------------------------------------------------------------------------
// Kernel 2: persistent recurrence. 16 WGs x 256 threads; WG j owns H columns
// [j*32, j*32+32). U_h slice lives in LDS (bf16, TRANSPOSED [col][k] with a
// +8 pad so each lane's B fragment is two conflict-skewed ds_load_b128).
// Per step: h_hat = tanh(xh_t + h @ U_h); h = mask ? h_hat*R_t : h.
// h state ping-pongs between two global f32 buffers with a grid sync per step.
// ---------------------------------------------------------------------------
__global__ __launch_bounds__(256) void k_recurrence(
    const float* __restrict__ U_h,
    const float* __restrict__ xh, const float* __restrict__ Rg,
    const int*   __restrict__ lengths,
    float* __restrict__ hA, float* __restrict__ hB,
    unsigned* __restrict__ counter)
{
    constexpr int US = Hn + 8;                 // padded k-stride per column
    __shared__ __bf16 Us[32 * US];             // ~33 KB

    const int j = blockIdx.x;
    const int tid = threadIdx.x;

    // Stage U_h column slice, transposed: Us[c][k] = U_h[k][j*32+c]
    for (int i = tid; i < Hn * 32; i += 256) {
        const int c = i & 31, k = i >> 5;      // consecutive tid -> coalesced read
        Us[c * US + k] = (__bf16)U_h[k * Hn + (j * 32 + c)];
    }
    // h0 = 0 on our column slice.
    for (int i = tid; i < Bn * 32; i += 256) {
        const int r = i >> 5, c = i & 31;
        hA[r * Hn + j * 32 + c] = 0.0f;
    }
    unsigned phase = 1;
    grid_sync(counter, NWG * phase); ++phase;

    const int lane = tid & 31;
    const int wave = tid >> 5;                 // 8 waves: 4 row x 2 col tiles
    const int mt = wave & 3;
    const int nt = wave >> 2;
    const int ml = lane & 15;
    const int kh = lane >> 4;
    const int mrow = mt * 16 + ml;             // A-fragment row (batch index)
    const int nl = nt * 16 + ml;               // local column 0..31
    const int n = j * 32 + nl;                 // global column
    const __bf16* us_row = &Us[nl * US];

    // per-lane epilogue row metadata (constant over t)
    int lenv[8];
#pragma unroll
    for (int r8 = 0; r8 < 8; ++r8) lenv[r8] = lengths[mt * 16 + r8 + kh * 8];

    for (int t = 0; t < Sn; ++t) {
        const float* hin  = (t & 1) ? hB : hA;
        float*       hout = (t & 1) ? hA : hB;

        v8f acc = {};
        const float* arow = hin + (long)mrow * Hn;
        for (int kc = 0; kc < Hn / 32; ++kc) {
            const int k0 = kc * 32 + kh * 8;
            const vf4 x0 = *(const vf4*)(arow + k0);
            const vf4 x1 = *(const vf4*)(arow + k0 + 4);
            const vf4 x2 = *(const vf4*)(arow + k0 + 16);
            const vf4 x3 = *(const vf4*)(arow + k0 + 20);
            v16bf a;
#pragma unroll
            for (int e = 0; e < 4; ++e) {
                a[e]      = (__bf16)x0[e];
                a[e + 4]  = (__bf16)x1[e];
                a[e + 8]  = (__bf16)x2[e];
                a[e + 12] = (__bf16)x3[e];
            }
            const v16bf bb = cat16(*(const v8bf*)(us_row + k0),
                                   *(const v8bf*)(us_row + k0 + 16));
            acc = __builtin_amdgcn_wmma_f32_16x16x32_bf16(
                false, a, false, bb, (short)0, acc, false, false);
        }

        const long base = (long)t * (Bn * Hn);
#pragma unroll
        for (int r8 = 0; r8 < 8; ++r8) {
            const int M = mt * 16 + r8 + kh * 8;   // batch row of this C elem
            const long idx = (long)M * Hn + n;
            const float hold = hin[idx];
            const float hh = tanhf(xh[base + idx] + acc[r8]);
            const float hn = hh * Rg[base + idx];
            hout[idx] = (t < lenv[r8]) ? hn : hold;
        }
        grid_sync(counter, NWG * phase); ++phase;
    }
    // S=512 even, last write (t=511, odd) landed in hA => final state in hA.
}

// ---------------------------------------------------------------------------
// Kernel 3: tiny FC head. logits[b,c] = h[b,:] @ fc_W[:,c] + fc_b[c]
// ---------------------------------------------------------------------------
__global__ __launch_bounds__(256) void k_fc(
    const float* __restrict__ h, const float* __restrict__ fc_W,
    const float* __restrict__ fc_b, float* __restrict__ out)
{
    const int tid = threadIdx.x;              // 256 = 64 batches x 4 classes
    const int b = tid >> 2, c = tid & 3;
    float acc = fc_b[c];
    for (int k = 0; k < Hn; ++k) acc += h[b * Hn + k] * fc_W[k * 4 + c];
    out[b * 4 + c] = acc;
}

// ---------------------------------------------------------------------------
extern "C" void kernel_launch(void* const* d_in, const int* in_sizes, int n_in,
                              void* d_out, int out_size, void* d_ws, size_t ws_size,
                              hipStream_t stream) {
    const int*   x       = (const int*)  d_in[0];
    const int*   lengths = (const int*)  d_in[1];
    const float* emb     = (const float*)d_in[2];
    const float* W_h     = (const float*)d_in[3];
    const float* b_Wh    = (const float*)d_in[4];
    const float* U_h     = (const float*)d_in[5];
    const float* b_h     = (const float*)d_in[6];
    const float* W_r     = (const float*)d_in[7];
    const float* b_Wr    = (const float*)d_in[8];
    const float* b_r     = (const float*)d_in[9];
    const float* fc_W    = (const float*)d_in[10];
    const float* fc_b    = (const float*)d_in[11];
    float* out = (float*)d_out;

    char* ws = (char*)d_ws;
    size_t off = 0;
    unsigned* counter = (unsigned*)(ws + off); off += 256;
    __bf16* WhT = (__bf16*)(ws + off); off += (size_t)En * Hn * sizeof(__bf16);
    __bf16* WrT = (__bf16*)(ws + off); off += (size_t)En * Hn * sizeof(__bf16);
    float* xh = (float*)(ws + off); off += (size_t)Sn * Bn * Hn * sizeof(float);
    float* Rg = (float*)(ws + off); off += (size_t)Sn * Bn * Hn * sizeof(float);
    float* hA = (float*)(ws + off); off += (size_t)Bn * Hn * sizeof(float);
    float* hB = (float*)(ws + off); off += (size_t)Bn * Hn * sizeof(float);

    k_prep<<<(En * Hn + 255) / 256, 256, 0, stream>>>(W_h, W_r, WhT, WrT, counter);
    k_embed_project<<<2048, 256, 0, stream>>>(
        x, emb, WhT, WrT, b_Wh, b_h, b_Wr, b_r, xh, Rg);
    k_recurrence<<<NWG, 256, 0, stream>>>(U_h, xh, Rg, lengths, hA, hB, counter);
    k_fc<<<1, 256, 0, stream>>>(hA, fc_W, fc_b, out);
}
